// InterBlock_55628416418477
// MI455X (gfx1250) — matrix operand order repaired and checked by reading, not compiled, not run on hardware
//
#include <hip/hip_runtime.h>
#include <math.h>

// ---------------- problem constants ----------------
#define B_   8
#define N_   128
#define D_   256
#define H_   8
#define HD_  32
#define FF_  512

typedef __attribute__((ext_vector_type(16))) _Float16 v16h;
typedef __attribute__((ext_vector_type(8)))  _Float16 v8h;
typedef __attribute__((ext_vector_type(8)))  float    v8f;
typedef __attribute__((ext_vector_type(4)))  unsigned v4u;
typedef __attribute__((ext_vector_type(8)))  int      v8i;
typedef __attribute__((ext_vector_type(4)))  int      v4i;

#if defined(__has_builtin)
#if __has_builtin(__builtin_amdgcn_tensor_load_to_lds) && __has_builtin(__builtin_amdgcn_s_wait_tensorcnt)
#define HAVE_TDM 1
#endif
#endif
#ifndef HAVE_TDM
#define HAVE_TDM 0
#endif

// ---------------- WMMA fragment helpers (wave32, 16x16x32 f16) ----------------
// A (16x32): lane 0-15 -> M=lane, elems 0..7 = K k0..k0+7, elems 8..15 = K k0+16..k0+23
//            lane 16-31 -> M=lane-16, same with K offset +8.
// B (32x16): pre-swizzled ("fragged") weights: 16 contiguous halves per (tile,lane).
// C/D (16x16 f32): vgpr r holds M = r + 8*(lane>=16), N = lane%16.

__device__ __forceinline__ v16h pack16(v8h lo, v8h hi) {
  v16h r;
#pragma unroll
  for (int i = 0; i < 8; ++i) { r[i] = lo[i]; r[i + 8] = hi[i]; }
  return r;
}

__device__ __forceinline__ v16h ld_a(const _Float16* base, int stride, int m0, int k0, int lane) {
  const _Float16* p = base + (size_t)(m0 + (lane & 15)) * stride + k0 + ((lane >> 4) << 3);
  return pack16(*(const v8h*)p, *(const v8h*)(p + 16));
}

__device__ __forceinline__ v16h ld_b(const _Float16* wf, int kt, int nt, int NT, int lane) {
  const _Float16* p = wf + ((size_t)((kt * NT + nt) * 32 + lane) << 4);
  return pack16(*(const v8h*)p, *(const v8h*)(p + 8));
}

#define WMMA_F16(acc, a, b) \
  (acc) = __builtin_amdgcn_wmma_f32_16x16x32_f16(false, (a), false, (b), (short)0, (acc), false, false)

__device__ __forceinline__ float wave_sum(float v) {
#pragma unroll
  for (int m = 16; m >= 1; m >>= 1) v += __shfl_xor(v, m, 32);
  return v;
}
__device__ __forceinline__ float wave_max(float v) {
#pragma unroll
  for (int m = 16; m >= 1; m >>= 1) v = fmaxf(v, __shfl_xor(v, m, 32));
  return v;
}

#if HAVE_TDM
// TDM 2-D tile load: `rows` rows x 256 f32, contiguous (stride 256), global -> LDS.
// D# per CDNA5 ISA 8.3/8.4: group0 = {flags, lds_addr, gaddr_lo, gaddr_hi|type=2},
// group1 = {data_size=4B, tensor_dim0=256, tensor_dim1=rows, tile_dim0=256,
//           tile_dim1=rows, dim0_stride=256}, groups 2/3 zero (2-D tensor).
// This toolchain exposes the 6-arg builtin:
//   (uint32x4 g0, int32x8 g1, int32x4 g2, int32x4 g3, int32x8 g4, int cpol)
__device__ __forceinline__ void tdm_load_rows(unsigned lds_off, const float* gsrc, int rows) {
  unsigned long long ga = (unsigned long long)(uintptr_t)gsrc;
  v4u g0 = { 1u,                                   // count=1 (valid user descriptor)
             lds_off,                              // lds_addr (bytes)
             (unsigned)(ga & 0xFFFFFFFFull),       // global_addr[31:0]
             (unsigned)((ga >> 32) & 0x01FFFFFFull) | (2u << 30) };  // addr[56:32] | type=2
  v8i g1 = { (int)(2u << 16),                      // data_size = 2 (4 bytes)
             (int)(256u << 16),                    // tensor_dim0[15:0]=256
             (int)((unsigned)rows << 16),          // dim0[31:16]=0 | tensor_dim1[15:0]=rows
             (int)(256u << 16),                    // dim1[31:16]=0 | tile_dim0=256
             rows,                                 // tile_dim1=rows | tile_dim2=0
             256,                                  // tensor_dim0_stride[31:0]=256
             0, 0 };
  v4i gz4 = { 0, 0, 0, 0 };
  v8i gz8 = { 0, 0, 0, 0, 0, 0, 0, 0 };
  __builtin_amdgcn_tensor_load_to_lds(g0, g1, gz4, gz4, gz8, 0);
}
#endif

// ---------------- K0: weight conversion f32 -> f16 B-fragment layout ----------------
__global__ void __launch_bounds__(256)
frag_weights_kernel(const float* __restrict__ W, int ldw, int NT,
                    _Float16* __restrict__ dst, int total) {
  int idx = blockIdx.x * 256 + threadIdx.x;
  if (idx >= total) return;
  int e = idx & 15;
  int l = (idx >> 4) & 31;
  int t = idx >> 9;
  int nt = t % NT, kt = t / NT;
  int k = kt * 32 + (e & 7) + ((e >> 3) << 4) + ((l >= 16) ? 8 : 0);
  int n = nt * 16 + (l & 15);
  dst[idx] = (_Float16)W[(size_t)k * ldw + n];
}

// ---------------- K1: node LN + fused QKV projection ----------------
__global__ void __launch_bounds__(256)
node_qkv_kernel(const float* __restrict__ atom,
                const float* __restrict__ lng, const float* __restrict__ lnb,
                const float* __restrict__ bq, const float* __restrict__ bk,
                const float* __restrict__ bv,
                const _Float16* __restrict__ wq, const _Float16* __restrict__ wk,
                const _Float16* __restrict__ wv,
                float* __restrict__ qf, float* __restrict__ kf, float* __restrict__ vf) {
  __shared__ _Float16 a_lds[16 * 256];
  int tid = threadIdx.x, lane = tid & 31, w = tid >> 5;
  int blk = blockIdx.x;
  int b = blk >> 3, i0 = (blk & 7) << 4;

  // LayerNorm 16 rows (2 per wave)
  for (int rr = w * 2; rr < w * 2 + 2; ++rr) {
    const float* row = atom + ((size_t)(b * N_ + i0 + rr)) * D_;
    int c0 = lane * 8;
    float x[8], s = 0.f, s2 = 0.f;
#pragma unroll
    for (int u = 0; u < 8; ++u) { x[u] = row[c0 + u]; s += x[u]; s2 += x[u] * x[u]; }
    s = wave_sum(s); s2 = wave_sum(s2);
    float mu = s * (1.f / D_);
    float var = s2 * (1.f / D_) - mu * mu;
    float rs = rsqrtf(var + 1e-5f);
#pragma unroll
    for (int u = 0; u < 8; ++u) {
      int c = c0 + u;
      a_lds[rr * 256 + c] = (_Float16)(((x[u] - mu) * rs) * lng[c] + lnb[c]);
    }
  }
  __syncthreads();

  const float qscale = 0.17677669529663687f;  // 1/sqrt(HD)
#pragma unroll
  for (int m = 0; m < 3; ++m) {   // fully unrolled -> compile-time pointer select
    const _Float16* wf = (m == 0) ? wq : (m == 1) ? wk : wv;
    const float* bias = (m == 0) ? bq : (m == 1) ? bk : bv;
    float* dst = (m == 0) ? qf : (m == 1) ? kf : vf;
    const float sc = (m == 0) ? qscale : 1.f;
    for (int itl = 0; itl < 2; ++itl) {
      int ntl = w + itl * 8;  // 0..15
      v8f acc = {};
#pragma unroll
      for (int kt = 0; kt < 8; ++kt) {
        v16h a = ld_a(a_lds, 256, 0, kt * 32, lane);
        v16h bb = ld_b(wf, kt, ntl, 16, lane);
        WMMA_F16(acc, a, bb);
      }
      int nlo = lane & 15, hi = lane >> 4;
      int n = ntl * 16 + nlo;
      float bn = bias[n];
#pragma unroll
      for (int r = 0; r < 8; ++r) {
        int row = i0 + r + 8 * hi;
        dst[((size_t)(b * N_ + row)) * D_ + n] = (acc[r] + bn) * sc;
      }
    }
  }
}

// ---------------- K2: TDM-staged pair LN + KE/VE WMMA GEMMs + attention ----------------
#define PSTR 264   // padded f16 row stride
// LDS: praw (2 x 64 x 256 f32) | p (128 x PSTR f16) | kv (128 x PSTR f16) | attn | qi | pre
#define PRAW_BYTES (2 * 64 * 256 * 4)
#define SMEM_ATTN (PRAW_BYTES + 128 * PSTR * 2 * 2 + (8 * 128 + 256 + 256) * 4)

__global__ void __launch_bounds__(256)
pair_attn_kernel(const float* __restrict__ pair,
                 const float* __restrict__ pg, const float* __restrict__ pbn,
                 const float* __restrict__ bke, const float* __restrict__ bve,
                 const _Float16* __restrict__ wke, const _Float16* __restrict__ wve,
                 const float* __restrict__ qf, const float* __restrict__ kf,
                 const float* __restrict__ vf,
                 const float* __restrict__ wvadd, const float* __restrict__ bvadd,
                 const float* __restrict__ pmask, const float* __restrict__ nmask,
                 const float* __restrict__ atom,
                 float* __restrict__ a1f, _Float16* __restrict__ a1h) {
  extern __shared__ __align__(16) char smem[];
  float*    praw   = (float*)smem;                          // 2 x (64 x 256)
  _Float16* p_lds  = (_Float16*)(smem + PRAW_BYTES);        // 128 x PSTR
  _Float16* kv_lds = p_lds + 128 * PSTR;                    // 128 x PSTR (KE then VE)
  float* attn = (float*)(smem + PRAW_BYTES + 128 * PSTR * 2 * 2);
  float* qi   = attn + 8 * 128;
  float* pre  = qi + 256;

  int tid = threadIdx.x, lane = tid & 31, w = tid >> 5;
  int blk = blockIdx.x;
  int b = blk >> 7, i = blk & 127;
  size_t rowbase = ((size_t)(b * N_) + i) * N_;

#if HAVE_TDM
  // Wave 0 kicks off both 64-row DMA chunks; first chunk gated by tensorcnt<=1
  // so rows 0..63 are LayerNorm'ed while rows 64..127 still stream from HBM.
  if (w == 0) {
    unsigned lds0 = (unsigned)(uintptr_t)(void*)praw;
    tdm_load_rows(lds0,                pair + rowbase * D_,         64);
    tdm_load_rows(lds0 + 64 * 256 * 4, pair + (rowbase + 64) * D_,  64);
    __builtin_amdgcn_s_wait_tensorcnt(1);
  }
#endif
  qi[tid] = qf[((size_t)(b * N_) + i) * D_ + tid];
  __syncthreads();

  // pair LayerNorm, 2 halves of 64 rows (8 rows per wave per half)
#pragma unroll
  for (int half = 0; half < 2; ++half) {
#if HAVE_TDM
    if (half == 1) {
      if (w == 0) __builtin_amdgcn_s_wait_tensorcnt(0);
      __syncthreads();
    }
#endif
    for (int jr = w * 8; jr < w * 8 + 8; ++jr) {
      int j = half * 64 + jr;
#if HAVE_TDM
      const float* row = praw + (size_t)half * (64 * 256) + (size_t)jr * 256;
#else
      const float* row = pair + (rowbase + j) * D_;
#endif
      int c0 = lane * 8;
      float x[8], s = 0.f, s2 = 0.f;
#pragma unroll
      for (int u = 0; u < 8; ++u) { x[u] = row[c0 + u]; s += x[u]; s2 += x[u] * x[u]; }
      s = wave_sum(s); s2 = wave_sum(s2);
      float mu = s * (1.f / D_);
      float var = s2 * (1.f / D_) - mu * mu;
      float rs = rsqrtf(var + 1e-5f);
#pragma unroll
      for (int u = 0; u < 8; ++u) {
        int c = c0 + u;
        p_lds[j * PSTR + c] = (_Float16)(((x[u] - mu) * rs) * pg[c] + pbn[c]);
      }
    }
  }
  __syncthreads();

  // KE' = p @ Wke + bke + k_node[j]   (128x256)
  for (int t = w; t < 128; t += 8) {
    int mt = t >> 4, nt = t & 15;
    v8f acc = {};
#pragma unroll
    for (int kt = 0; kt < 8; ++kt) {
      v16h a = ld_a(p_lds, PSTR, mt * 16, kt * 32, lane);
      v16h bb = ld_b(wke, kt, nt, 16, lane);
      WMMA_F16(acc, a, bb);
    }
    int nlo = lane & 15, hi = lane >> 4, c = nt * 16 + nlo;
    float bc = bke[c];
#pragma unroll
    for (int r = 0; r < 8; ++r) {
      int j = mt * 16 + r + 8 * hi;
      kv_lds[j * PSTR + c] = (_Float16)(acc[r] + bc + kf[((size_t)(b * N_) + j) * D_ + c]);
    }
  }
  __syncthreads();

  // logits + softmax: wave w == head h, each lane 4 j's (vectorized v8h LDS reads)
  {
    int h = w;
    float lg[4];
#pragma unroll
    for (int jj = 0; jj < 4; ++jj) {
      int j = jj * 32 + lane;
      const v8h* kr = (const v8h*)(kv_lds + j * PSTR + h * 32);
      float dt = 0.f;
#pragma unroll
      for (int ch = 0; ch < 4; ++ch) {
        v8h cc = kr[ch];
#pragma unroll
        for (int u = 0; u < 8; ++u) dt += qi[h * 32 + ch * 8 + u] * (float)cc[u];
      }
      float m = pmask[rowbase + j];
      lg[jj] = dt + (1.f - m) * -1e6f;
    }
    float mx = fmaxf(fmaxf(lg[0], lg[1]), fmaxf(lg[2], lg[3]));
    mx = wave_max(mx);
    float sm = 0.f;
#pragma unroll
    for (int jj = 0; jj < 4; ++jj) { lg[jj] = __expf(lg[jj] - mx); sm += lg[jj]; }
    sm = wave_sum(sm);
    float inv = 1.f / sm;
#pragma unroll
    for (int jj = 0; jj < 4; ++jj) attn[h * 128 + jj * 32 + lane] = lg[jj] * inv;
  }
  __syncthreads();

  // VE' = p @ Wve + bve + v_node[j]  (reuse kv_lds)
  for (int t = w; t < 128; t += 8) {
    int mt = t >> 4, nt = t & 15;
    v8f acc = {};
#pragma unroll
    for (int kt = 0; kt < 8; ++kt) {
      v16h a = ld_a(p_lds, PSTR, mt * 16, kt * 32, lane);
      v16h bb = ld_b(wve, kt, nt, 16, lane);
      WMMA_F16(acc, a, bb);
    }
    int nlo = lane & 15, hi = lane >> 4, c = nt * 16 + nlo;
    float bc = bve[c];
#pragma unroll
    for (int r = 0; r < 8; ++r) {
      int j = mt * 16 + r + 8 * hi;
      kv_lds[j * PSTR + c] = (_Float16)(acc[r] + bc + vf[((size_t)(b * N_) + j) * D_ + c]);
    }
  }
  __syncthreads();

  // s = v_i + sum_j attn * VE'   (linearity of Wvadd + softmax normalization)
  {
    int h = w, d = lane;
    float s = vf[((size_t)(b * N_) + i) * D_ + h * 32 + d];
    const float* at = attn + h * 128;
    for (int j = 0; j < 128; ++j) s += at[j] * (float)kv_lds[j * PSTR + h * 32 + d];
    pre[h * 32 + d] = s;
  }
  __syncthreads();
  {
    int h = w, dc = lane;
    float o = bvadd[dc];
#pragma unroll
    for (int d2 = 0; d2 < 32; ++d2) o += pre[h * 32 + d2] * wvadd[d2 * 32 + dc];
    float nm = nmask[b * N_ + i];
    size_t idx = ((size_t)(b * N_) + i) * D_ + h * 32 + dc;
    float a1 = atom[idx] + o * nm;
    a1f[idx] = a1;
    a1h[idx] = (_Float16)a1;
  }
}

// ---------------- K3: FFN (GEMM-GELU-GEMM + residual), 16-row tiles ----------------
__global__ void __launch_bounds__(256)
ffn_kernel(const _Float16* __restrict__ a1h, const float* __restrict__ a1f,
           const _Float16* __restrict__ w1, const float* __restrict__ b1,
           const _Float16* __restrict__ w2, const float* __restrict__ b2,
           float* __restrict__ outa, _Float16* __restrict__ a2h) {
  __shared__ _Float16 h1[16 * 520];
  int tid = threadIdx.x, lane = tid & 31, w = tid >> 5;
  int blk = blockIdx.x;
  int b = blk >> 3, i0 = (blk & 7) << 4;
  const _Float16* abase = a1h + ((size_t)(b * N_) + i0) * D_;

  for (int it = 0; it < 4; ++it) {
    int nt = w + it * 8;  // 0..31 (FF/16)
    v8f acc = {};
#pragma unroll
    for (int kt = 0; kt < 8; ++kt) {
      v16h a = ld_a(abase, D_, 0, kt * 32, lane);
      v16h bb = ld_b(w1, kt, nt, 32, lane);
      WMMA_F16(acc, a, bb);
    }
    int nlo = lane & 15, hi = lane >> 4, n = nt * 16 + nlo;
    float bn = b1[n];
#pragma unroll
    for (int r = 0; r < 8; ++r) {
      float x = acc[r] + bn;
      float gl = 0.5f * x * (1.f + erff(x * 0.70710678118654752f));  // exact GELU
      h1[(r + 8 * hi) * 520 + n] = (_Float16)gl;
    }
  }
  __syncthreads();

  for (int it = 0; it < 2; ++it) {
    int nt = w + it * 8;  // 0..15
    v8f acc = {};
#pragma unroll
    for (int kt = 0; kt < 16; ++kt) {
      v16h a = ld_a(h1, 520, 0, kt * 32, lane);
      v16h bb = ld_b(w2, kt, nt, 16, lane);
      WMMA_F16(acc, a, bb);
    }
    int nlo = lane & 15, hi = lane >> 4, n = nt * 16 + nlo;
    float bn = b2[n];
#pragma unroll
    for (int r = 0; r < 8; ++r) {
      int row = i0 + r + 8 * hi;
      size_t idx = ((size_t)(b * N_) + row) * D_ + n;
      float v = a1f[idx] + acc[r] + bn;
      outa[idx] = v;
      a2h[idx] = (_Float16)v;
    }
  }
}

// ---------------- K4: cm_w1 separable halves: r = a2@W1[0:D], c = a2@W1[D:2D] ----------------
__global__ void __launch_bounds__(256)
posprep_kernel(const _Float16* __restrict__ a2h,
               const _Float16* __restrict__ w1r, const _Float16* __restrict__ w1c,
               _Float16* __restrict__ rc) {
  int tid = threadIdx.x, lane = tid & 31, w = tid >> 5;
  int blk = blockIdx.x;
  int b = blk >> 3, i0 = (blk & 7) << 4;
  const _Float16* abase = a2h + ((size_t)(b * N_) + i0) * D_;

#pragma unroll
  for (int half = 0; half < 2; ++half) {   // compile-time weight select
    const _Float16* wf = half ? w1c : w1r;
    for (int itl = 0; itl < 2; ++itl) {
      int ntl = w + itl * 8;  // 0..15
      v8f acc = {};
#pragma unroll
      for (int kt = 0; kt < 8; ++kt) {
        v16h a = ld_a(abase, D_, 0, kt * 32, lane);
        v16h bb = ld_b(wf, kt, ntl, 16, lane);
        WMMA_F16(acc, a, bb);
      }
      int nlo = lane & 15, hi = lane >> 4;
      int n = half * 256 + ntl * 16 + nlo;  // 0..511
#pragma unroll
      for (int r = 0; r < 8; ++r) {
        int row = i0 + r + 8 * hi;
        rc[((size_t)(b * N_) + row) * 512 + n] = (_Float16)acc[r];
      }
    }
  }
}

// ---------------- K5: PosUpdate per (b,i): build h1 in LDS, h2 GEMM, gate, coord reduce ----------------
#define SMEM_POS (128 * PSTR * 2 + 128 * 4 + (256 * 3 + 768) * 4)

__global__ void __launch_bounds__(256)
pos_update_kernel(const float* __restrict__ pair,
                  const float* __restrict__ ddw, const float* __restrict__ ddb,
                  const float* __restrict__ cmw1, const float* __restrict__ cmb1,
                  const float* __restrict__ cmb2, const float* __restrict__ cmw3,
                  const _Float16* __restrict__ wcm2,
                  const _Float16* __restrict__ rc,
                  const float* __restrict__ pmask, const float* __restrict__ cdiff,
                  const float* __restrict__ pos, float* __restrict__ outpos) {
  extern __shared__ __align__(16) char smem[];
  _Float16* h1 = (_Float16*)smem;                    // 128 x PSTR
  float* gate = (float*)(smem + 128 * PSTR * 2);     // 128
  float* ri   = gate + 128;                          // 256
  float* w3s  = ri + 256;                            // 256
  float* b1s  = w3s + 256;                           // 256
  float* eW   = b1s + 256;                           // 3 x 256

  int tid = threadIdx.x, lane = tid & 31, w = tid >> 5;
  int blk = blockIdx.x;
  int b = blk >> 7, i = blk & 127;
  size_t rowbase = ((size_t)(b * N_) + i) * N_;

  ri[tid]  = (float)rc[((size_t)(b * N_) + i) * 512 + tid];
  w3s[tid] = cmw3[tid];
  b1s[tid] = cmb1[tid];
#pragma unroll
  for (int t = 0; t < 3; ++t) eW[t * 256 + tid] = cmw1[(size_t)(512 + t) * 256 + tid];
  if (tid < 128) gate[tid] = 0.f;
  __syncthreads();

  // h1[j][c] = silu(r_i[c] + c_j[c] + edge3·W1e[:,c] + b1[c])
  for (int j = w * 16; j < w * 16 + 16; ++j) {
    const float* prow = pair + (rowbase + j) * D_;
    int c0 = lane * 8;
    float e0 = 0.f, e1 = 0.f, e2 = 0.f;
#pragma unroll
    for (int u = 0; u < 8; ++u) {
      float x = prow[c0 + u];
      const float* dw = ddw + (size_t)(c0 + u) * 3;
      e0 += x * dw[0]; e1 += x * dw[1]; e2 += x * dw[2];
    }
    e0 = wave_sum(e0) + ddb[0];
    e1 = wave_sum(e1) + ddb[1];
    e2 = wave_sum(e2) + ddb[2];
    const _Float16* crow = rc + ((size_t)(b * N_) + j) * 512 + 256;
#pragma unroll
    for (int u = 0; u < 8; ++u) {
      int c = c0 + u;
      float x = ri[c] + (float)crow[c] + e0 * eW[c] + e1 * eW[256 + c] + e2 * eW[512 + c] + b1s[c];
      float s = x * (1.f / (1.f + __expf(-x)));  // silu
      h1[j * PSTR + c] = (_Float16)s;
    }
  }
  __syncthreads();

  // h2 = silu(h1 @ cm_w2 + b2); gate[j] += h2·cm_w3 (fused via LDS atomics)
  for (int t = w; t < 128; t += 8) {
    int mt = t >> 4, nt = t & 15;
    v8f acc = {};
#pragma unroll
    for (int kt = 0; kt < 8; ++kt) {
      v16h a = ld_a(h1, PSTR, mt * 16, kt * 32, lane);
      v16h bb = ld_b(wcm2, kt, nt, 16, lane);
      WMMA_F16(acc, a, bb);
    }
    int nlo = lane & 15, hi = lane >> 4, n = nt * 16 + nlo;
    float bn = cmb2[n], wn = w3s[n];
#pragma unroll
    for (int r = 0; r < 8; ++r) {
      int j = mt * 16 + r + 8 * hi;
      float x = acc[r] + bn;
      float s = x * (1.f / (1.f + __expf(-x)));
      atomicAdd(&gate[j], s * wn);
    }
  }
  __syncthreads();

  if (w == 0) {
    float t0 = 0.f, t1 = 0.f, t2 = 0.f;
#pragma unroll
    for (int jj = 0; jj < 4; ++jj) {
      int j = jj * 32 + lane;
      float g = gate[j] * pmask[rowbase + j];
      const float* cd = cdiff + (rowbase + j) * 3;
      t0 += cd[0] * g; t1 += cd[1] * g; t2 += cd[2] * g;
    }
    t0 = wave_sum(t0); t1 = wave_sum(t1); t2 = wave_sum(t2);
    if (lane == 0) {
      const float* pp = pos + ((size_t)(b * N_) + i) * 3;
      float* op = outpos + ((size_t)(b * N_) + i) * 3;
      op[0] = pp[0] + t0; op[1] = pp[1] + t1; op[2] = pp[2] + t2;
    }
  }
}

// ---------------- host ----------------
extern "C" void kernel_launch(void* const* d_in, const int* in_sizes, int n_in,
                              void* d_out, int out_size, void* d_ws, size_t ws_size,
                              hipStream_t stream) {
  const float* atom   = (const float*)d_in[0];
  const float* pair   = (const float*)d_in[1];
  const float* pos    = (const float*)d_in[2];
  const float* cdiff  = (const float*)d_in[3];
  const float* nmask  = (const float*)d_in[4];
  const float* pmask  = (const float*)d_in[5];
  const float* nlng   = (const float*)d_in[6];
  const float* nlnb   = (const float*)d_in[7];
  const float* plng   = (const float*)d_in[8];
  const float* plnb   = (const float*)d_in[9];
  const float* Wq  = (const float*)d_in[10]; const float* bq  = (const float*)d_in[11];
  const float* Wk  = (const float*)d_in[12]; const float* bk  = (const float*)d_in[13];
  const float* Wv  = (const float*)d_in[14]; const float* bv  = (const float*)d_in[15];
  const float* Wke = (const float*)d_in[16]; const float* bke = (const float*)d_in[17];
  const float* Wve = (const float*)d_in[18]; const float* bve = (const float*)d_in[19];
  const float* Wvadd = (const float*)d_in[20]; const float* bvadd = (const float*)d_in[21];
  const float* fw1 = (const float*)d_in[22]; const float* fb1 = (const float*)d_in[23];
  const float* fw2 = (const float*)d_in[24]; const float* fb2 = (const float*)d_in[25];
  const float* ddw = (const float*)d_in[26]; const float* ddb = (const float*)d_in[27];
  const float* cmw1 = (const float*)d_in[28]; const float* cmb1 = (const float*)d_in[29];
  const float* cmw2 = (const float*)d_in[30]; const float* cmb2 = (const float*)d_in[31];
  const float* cmw3 = (const float*)d_in[32];

  char* ws = (char*)d_ws;
  _Float16* wq_f   = (_Float16*)(ws + 0);
  _Float16* wk_f   = (_Float16*)(ws + 131072);
  _Float16* wv_f   = (_Float16*)(ws + 262144);
  _Float16* wke_f  = (_Float16*)(ws + 393216);
  _Float16* wve_f  = (_Float16*)(ws + 524288);
  _Float16* wf1_f  = (_Float16*)(ws + 655360);
  _Float16* wf2_f  = (_Float16*)(ws + 917504);
  _Float16* w1r_f  = (_Float16*)(ws + 1179648);
  _Float16* w1c_f  = (_Float16*)(ws + 1310720);
  _Float16* wcm2_f = (_Float16*)(ws + 1441792);
  float*    qf  = (float*)(ws + 1572864);
  float*    kf  = (float*)(ws + 2621440);
  float*    vf  = (float*)(ws + 3670016);
  float*    a1f = (float*)(ws + 4718592);
  _Float16* a1h = (_Float16*)(ws + 5767168);
  _Float16* a2h = (_Float16*)(ws + 6291456);
  _Float16* rc  = (_Float16*)(ws + 6815744);

  float* outa   = (float*)d_out;
  float* outpos = (float*)d_out + (size_t)B_ * N_ * D_;

  (void)hipFuncSetAttribute(reinterpret_cast<const void*>(pair_attn_kernel),
                            hipFuncAttributeMaxDynamicSharedMemorySize, SMEM_ATTN);
  (void)hipFuncSetAttribute(reinterpret_cast<const void*>(pos_update_kernel),
                            hipFuncAttributeMaxDynamicSharedMemorySize, SMEM_POS);

  // K0: fragment all GEMM weights into f16 WMMA-B layout
  frag_weights_kernel<<<256, 256, 0, stream>>>(Wq,  256, 16, wq_f,  65536);
  frag_weights_kernel<<<256, 256, 0, stream>>>(Wk,  256, 16, wk_f,  65536);
  frag_weights_kernel<<<256, 256, 0, stream>>>(Wv,  256, 16, wv_f,  65536);
  frag_weights_kernel<<<256, 256, 0, stream>>>(Wke, 256, 16, wke_f, 65536);
  frag_weights_kernel<<<256, 256, 0, stream>>>(Wve, 256, 16, wve_f, 65536);
  frag_weights_kernel<<<512, 256, 0, stream>>>(fw1, 512, 32, wf1_f, 131072);
  frag_weights_kernel<<<512, 256, 0, stream>>>(fw2, 256, 16, wf2_f, 131072);
  frag_weights_kernel<<<256, 256, 0, stream>>>(cmw1,             256, 16, w1r_f, 65536);
  frag_weights_kernel<<<256, 256, 0, stream>>>(cmw1 + 256 * 256, 256, 16, w1c_f, 65536);
  frag_weights_kernel<<<256, 256, 0, stream>>>(cmw2, 256, 16, wcm2_f, 65536);

  // K1: node LN + QKV
  node_qkv_kernel<<<B_ * N_ / 16, 256, 0, stream>>>(atom, nlng, nlnb, bq, bk, bv,
                                                    wq_f, wk_f, wv_f, qf, kf, vf);
  // K2: TDM-staged pair LN + KE/VE + attention -> atom1
  pair_attn_kernel<<<B_ * N_, 256, SMEM_ATTN, stream>>>(pair, plng, plnb, bke, bve,
                                                        wke_f, wve_f, qf, kf, vf,
                                                        Wvadd, bvadd, pmask, nmask,
                                                        atom, a1f, a1h);
  // K3: FFN -> atom2 (writes d_out atom part)
  ffn_kernel<<<B_ * N_ / 16, 256, 0, stream>>>(a1h, a1f, wf1_f, fb1, wf2_f, fb2, outa, a2h);
  // K4: separable cm_w1 row/col projections
  posprep_kernel<<<B_ * N_ / 16, 256, 0, stream>>>(a2h, w1r_f, w1c_f, rc);
  // K5: PosUpdate -> new_pos
  pos_update_kernel<<<B_ * N_, 256, SMEM_POS, stream>>>(pair, ddw, ddb, cmw1, cmb1,
                                                        cmb2, cmw3, wcm2_f, rc,
                                                        pmask, cdiff, pos, outpos);
}
